// HRoPEv6Model_10153302687758
// MI455X (gfx1250) — matrix operand order repaired and checked by reading, not compiled, not run on hardware
//
#include <hip/hip_runtime.h>
#include <hip/hip_bf16.h>
#include <math.h>

// ---------------------------------------------------------------------------
// HRoPEv6 hierarchical transformer forward, CDNA5 (gfx1250), wave32 + WMMA.
// GEMMs: bf16 A-side activations, async-to-LDS A staging (ASYNCcnt),
// register-staged transposed B (v_cvt_pk_bf16_f32 packing), double-buffered,
// 128x64 block tile, 8x v_wmma_f32_16x16x32_bf16 per K-step per wave.
// ---------------------------------------------------------------------------

typedef __attribute__((ext_vector_type(16))) __bf16 v16bf;
typedef __attribute__((ext_vector_type(8)))  __bf16 v8bf;
typedef __attribute__((ext_vector_type(8)))  float  v8f;
typedef __attribute__((address_space(3)))    __bf16 lds_bf16;

#define D_    1024
#define H_    16
#define HD_   64
#define FFN_  4096
#define T_    2048
#define NS_   64
#define NP_   8
#define EPS_  1e-6f

#define LDS_STRIDE 40   // bf16 elements per LDS row (80 B, 16 B aligned, padded)

#define WAIT_ASYNC(n) asm volatile("s_wait_asynccnt " #n ::: "memory")

__device__ inline void async_copy_b128(lds_bf16* dst, const __bf16* src) {
  // GLOBAL_LOAD_ASYNC_TO_LDS_B128: VDST = LDS byte address (VGPR),
  // VADDR = 64-bit global address (VGPR pair). Tracked by ASYNCcnt.
  asm volatile("global_load_async_to_lds_b128 %0, %1, off"
               :: "v"(dst), "v"(src) : "memory");
}

__device__ inline unsigned pack_bf16x2(float a, float b) {
  // Single-instruction packed convert: dst[15:0]=bf16(a), dst[31:16]=bf16(b).
  unsigned u;
  asm("v_cvt_pk_bf16_f32 %0, %1, %2" : "=v"(u) : "v"(a), "v"(b));
  return u;
}

// ------------------------------ embed --------------------------------------
__global__ void k_embed(const float* __restrict__ emb, const int* __restrict__ ids,
                        float* __restrict__ x) {
  int t = blockIdx.x;
  int row = ids[t];
  const float4* src = (const float4*)(emb + (size_t)row * D_);
  float4* dst = (float4*)(x + (size_t)t * D_);
  dst[threadIdx.x] = src[threadIdx.x];
}

// ------------------------------ rms norms ----------------------------------
__global__ void k_rms_bf16(const float* __restrict__ x, const float* __restrict__ w,
                           __bf16* __restrict__ y, int D) {
  int r = blockIdx.x;
  const float* xr = x + (size_t)r * D;
  float ss = 0.f;
  for (int i = threadIdx.x; i < D; i += blockDim.x) { float v = xr[i]; ss += v * v; }
  __shared__ float red[256];
  red[threadIdx.x] = ss; __syncthreads();
  for (int s = blockDim.x >> 1; s > 0; s >>= 1) {
    if ((int)threadIdx.x < s) red[threadIdx.x] += red[threadIdx.x + s];
    __syncthreads();
  }
  float inv = rsqrtf(red[0] / (float)D + EPS_);
  for (int i = threadIdx.x; i < D; i += blockDim.x)
    y[(size_t)r * D + i] = (__bf16)(xr[i] * inv * w[i]);
}

__global__ void k_rms_f32(const float* __restrict__ x, const float* __restrict__ w,
                          float* __restrict__ y, int D) {
  int r = blockIdx.x;
  const float* xr = x + (size_t)r * D;
  float ss = 0.f;
  for (int i = threadIdx.x; i < D; i += blockDim.x) { float v = xr[i]; ss += v * v; }
  __shared__ float red[256];
  red[threadIdx.x] = ss; __syncthreads();
  for (int s = blockDim.x >> 1; s > 0; s >>= 1) {
    if ((int)threadIdx.x < s) red[threadIdx.x] += red[threadIdx.x + s];
    __syncthreads();
  }
  float inv = rsqrtf(red[0] / (float)D + EPS_);
  for (int i = threadIdx.x; i < D; i += blockDim.x)
    y[(size_t)r * D + i] = xr[i] * inv * w[i];
}

// ------------------------------ WMMA GEMM ----------------------------------
// C[M,N] = epi(A[M,K] @ B[K,N]); A bf16 row-major, B f32 row-major.
// Block: 128 threads (4 waves), 128x64 tile, K-step 32, double-buffered LDS.
// A slab: async copy global->LDS (ASYNCcnt). B slab: regs + transpose + cvt.
// Each wave: 32 rows -> 2 A frags x 4 B frags = 8 WMMA per K-step.
// epi: 0 = none (f32 out), 1 = += Cres (f32 out), 2 = exact GELU (bf16 out).
__global__ void k_gemm(const __bf16* __restrict__ A, const float* __restrict__ B,
                       const float* __restrict__ Cres, void* __restrict__ Cout,
                       int M, int N, int K, int epi) {
  __shared__ unsigned ldsA[2][128 * (LDS_STRIDE / 2)];  // bf16 [m][k], padded
  __shared__ unsigned ldsB[2][64  * (LDS_STRIDE / 2)];  // bf16 [n][k] (transposed)

  int t    = threadIdx.x;
  int wave = t >> 5;
  int lane = t & 31;
  int m0   = blockIdx.y * 128;
  int n0   = blockIdx.x * 64;

  // staging work split
  int nB  = t & 63;            // B: column handled by this thread
  int kB0 = (t >> 6) * 2;      // B: base k (0 or 2), step 4
  int mAs = t;                 // A: local row (0..127), full 32-k row per thread
  int mgA = m0 + mAs; if (mgA >= M) mgA = M - 1;        // clamp ragged M

  // compute-side indices (ISA m/n mapping; contiguous k = 16*(lane/16) + s)
  int mloc0 = wave * 32 + (lane & 15);
  int mloc1 = mloc0 + 16;
  int koffA = (lane & 16) ? 16 : 0;
  int koffB = (lane & 16) ? 16 : 0;
  int nl    = lane & 15;

  float rB[16];
  int nk = K >> 5;

  auto asyncA = [&](int k0, int buf) {
    const __bf16* gp = A + (size_t)mgA * K + k0;
    __bf16* lp = (__bf16*)&ldsA[buf][0] + mAs * LDS_STRIDE;
    async_copy_b128((lds_bf16*)(lp),      gp);
    async_copy_b128((lds_bf16*)(lp +  8), gp +  8);
    async_copy_b128((lds_bf16*)(lp + 16), gp + 16);
    async_copy_b128((lds_bf16*)(lp + 24), gp + 24);
  };
  auto gloadB = [&](int k0) {
#pragma unroll
    for (int it = 0; it < 8; ++it) {
      int k = kB0 + 4 * it;
      rB[2 * it]     = B[(size_t)(k0 + k)     * N + n0 + nB];
      rB[2 * it + 1] = B[(size_t)(k0 + k + 1) * N + n0 + nB];
    }
  };
  auto stageB = [&](int buf) {
#pragma unroll
    for (int it = 0; it < 8; ++it) {
      int k = kB0 + 4 * it;
      ldsB[buf][nB * (LDS_STRIDE / 2) + (k >> 1)] = pack_bf16x2(rB[2 * it], rB[2 * it + 1]);
    }
  };

  v8f a0b0 = {}, a0b1 = {}, a0b2 = {}, a0b3 = {};
  v8f a1b0 = {}, a1b1 = {}, a1b2 = {}, a1b3 = {};

  asyncA(0, 0);
  gloadB(0);
  stageB(0);
  if (nk > 1) { asyncA(32, 1); gloadB(32); WAIT_ASYNC(4); }
  else        { WAIT_ASYNC(0); }
  __syncthreads();

  for (int i = 0; i < nk; ++i) {
    int cur = i & 1;
    const __bf16* baseA = (const __bf16*)&ldsA[cur][0];
    const __bf16* baseB = (const __bf16*)&ldsB[cur][0];

    // two A fragments (rows mloc0, mloc1): aligned b128 LDS loads + shuffles
    const __bf16* pa0 = baseA + mloc0 * LDS_STRIDE + koffA;
    v8bf a0lo = *(const v8bf*)(pa0);
    v8bf a0hi = *(const v8bf*)(pa0 + 8);
    v16bf a0 = __builtin_shufflevector(a0lo, a0hi, 0,1,2,3,4,5,6,7,8,9,10,11,12,13,14,15);
    const __bf16* pa1 = baseA + mloc1 * LDS_STRIDE + koffA;
    v8bf a1lo = *(const v8bf*)(pa1);
    v8bf a1hi = *(const v8bf*)(pa1 + 8);
    v16bf a1 = __builtin_shufflevector(a1lo, a1hi, 0,1,2,3,4,5,6,7,8,9,10,11,12,13,14,15);

    {
      const __bf16* pb = baseB + (nl +  0) * LDS_STRIDE + koffB;
      v8bf blo = *(const v8bf*)(pb); v8bf bhi = *(const v8bf*)(pb + 8);
      v16bf b = __builtin_shufflevector(blo, bhi, 0,1,2,3,4,5,6,7,8,9,10,11,12,13,14,15);
      a0b0 = __builtin_amdgcn_wmma_f32_16x16x32_bf16(false, a0, false, b, (short)0, a0b0, false, false);
      a1b0 = __builtin_amdgcn_wmma_f32_16x16x32_bf16(false, a1, false, b, (short)0, a1b0, false, false);
    }
    {
      const __bf16* pb = baseB + (nl + 16) * LDS_STRIDE + koffB;
      v8bf blo = *(const v8bf*)(pb); v8bf bhi = *(const v8bf*)(pb + 8);
      v16bf b = __builtin_shufflevector(blo, bhi, 0,1,2,3,4,5,6,7,8,9,10,11,12,13,14,15);
      a0b1 = __builtin_amdgcn_wmma_f32_16x16x32_bf16(false, a0, false, b, (short)0, a0b1, false, false);
      a1b1 = __builtin_amdgcn_wmma_f32_16x16x32_bf16(false, a1, false, b, (short)0, a1b1, false, false);
    }
    {
      const __bf16* pb = baseB + (nl + 32) * LDS_STRIDE + koffB;
      v8bf blo = *(const v8bf*)(pb); v8bf bhi = *(const v8bf*)(pb + 8);
      v16bf b = __builtin_shufflevector(blo, bhi, 0,1,2,3,4,5,6,7,8,9,10,11,12,13,14,15);
      a0b2 = __builtin_amdgcn_wmma_f32_16x16x32_bf16(false, a0, false, b, (short)0, a0b2, false, false);
      a1b2 = __builtin_amdgcn_wmma_f32_16x16x32_bf16(false, a1, false, b, (short)0, a1b2, false, false);
    }
    {
      const __bf16* pb = baseB + (nl + 48) * LDS_STRIDE + koffB;
      v8bf blo = *(const v8bf*)(pb); v8bf bhi = *(const v8bf*)(pb + 8);
      v16bf b = __builtin_shufflevector(blo, bhi, 0,1,2,3,4,5,6,7,8,9,10,11,12,13,14,15);
      a0b3 = __builtin_amdgcn_wmma_f32_16x16x32_bf16(false, a0, false, b, (short)0, a0b3, false, false);
      a1b3 = __builtin_amdgcn_wmma_f32_16x16x32_bf16(false, a1, false, b, (short)0, a1b3, false, false);
    }

    if (i + 1 < nk) {
      __syncthreads();                   // everyone done reading lds[*][cur]
      stageB(cur ^ 1);                   // stage B slab i+1
      if (i + 2 < nk) {
        gloadB((i + 2) << 5);            // issue B slab i+2 global loads
        asyncA((i + 2) << 5, cur);       // async A slab i+2 -> ldsA[cur]
        WAIT_ASYNC(4);                   // in-order: slab i+1 landed
      } else {
        WAIT_ASYNC(0);                   // last slab landed
      }
      __syncthreads();                   // staged data visible
    }
  }

  // C/D layout: VGPR r -> row base + r + 8*(lane>=16); lane&15 -> column
  int mhi = (lane & 16) ? 8 : 0;
  int mbase0 = m0 + wave * 32 + mhi;
  int mbase1 = mbase0 + 16;
#pragma unroll
  for (int r = 0; r < 8; ++r) {
#pragma unroll
    for (int half = 0; half < 2; ++half) {
      int m = (half ? mbase1 : mbase0) + r;
      if (m < M) {
        size_t off = (size_t)m * N + n0 + nl;
        float vv[4];
        if (half) { vv[0] = a1b0[r]; vv[1] = a1b1[r]; vv[2] = a1b2[r]; vv[3] = a1b3[r]; }
        else      { vv[0] = a0b0[r]; vv[1] = a0b1[r]; vv[2] = a0b2[r]; vv[3] = a0b3[r]; }
#pragma unroll
        for (int j = 0; j < 4; ++j) {
          float val = vv[j];
          if (epi == 2) {
            val = 0.5f * val * (1.0f + erff(val * 0.70710678118654752f));
            ((__bf16*)Cout)[off + 16 * j] = (__bf16)val;
          } else {
            if (epi == 1) val += Cres[off + 16 * j];
            ((float*)Cout)[off + 16 * j] = val;
          }
        }
      }
    }
  }
}

// ------------------- per-head RMS norm + RoPE (q and k) --------------------
__global__ void k_qkrope(float* __restrict__ q, const float* __restrict__ w,
                         const int* __restrict__ pos, float base) {
  int r = blockIdx.x;
  int h = blockIdx.y;
  float* p = q + (size_t)r * D_ + h * HD_;
  int d = threadIdx.x;                       // 0..63
  float v = p[d];
  __shared__ float red[64];
  __shared__ float nv[64];
  red[d] = v * v; __syncthreads();
  for (int s = 32; s > 0; s >>= 1) { if (d < s) red[d] += red[d + s]; __syncthreads(); }
  float inv = rsqrtf(red[0] / (float)HD_ + EPS_);
  nv[d] = v * inv * w[d];
  __syncthreads();
  if (d < 32) {
    float fpos = (float)pos[r];
    float invf = powf(base, -(float)d / 32.0f);
    float f = fpos * invf;
    float sn, cs; sincosf(f, &sn, &cs);
    float x1 = nv[d], x2 = nv[d + 32];
    p[d]      = x1 * cs - x2 * sn;
    p[d + 32] = x1 * sn + x2 * cs;
  }
}

// ------------- relu^2 attention, contiguous group mask (self) --------------
// 64 threads; sliced dot products, 3 barriers per block; bf16 output.
__global__ void k_attn(const float* __restrict__ q, const float* __restrict__ k,
                       const float* __restrict__ v, __bf16* __restrict__ o, int gsize) {
  int r = blockIdx.x;
  int h = blockIdx.y;
  int g = r / gsize, j0 = g * gsize;
  int t = threadIdx.x;                       // 0..63
  int slices = HD_ / gsize;                  // gsize 8 -> 8, gsize 32 -> 2
  int jj = t % gsize;
  int sl = t / gsize;
  int w  = HD_ / slices;
  __shared__ float ps[64];
  __shared__ float wj[32];
  __shared__ float denom_s;
  const float* qrow = q + (size_t)r * D_ + h * HD_;
  const float* krow = k + (size_t)(j0 + jj) * D_ + h * HD_;
  float acc = 0.f;
  for (int i = 0; i < w; ++i) { int d = sl * w + i; acc += qrow[d] * krow[d]; }
  ps[t] = acc;
  __syncthreads();
  if (t < gsize) {
    float s = 0.f;
    for (int s2 = 0; s2 < slices; ++s2) s += ps[t + gsize * s2];
    float sc = s * 0.125f;                   // 1/sqrt(64)
    sc = sc > 0.f ? sc * sc : 0.f;
    wj[t] = sc;
  }
  __syncthreads();
  if (t == 0) {
    float s = 0.f;
    for (int j = 0; j < gsize; ++j) s += wj[j];
    denom_s = s < EPS_ ? EPS_ : s;
  }
  __syncthreads();
  float oa = 0.f;
  for (int j = 0; j < gsize; ++j)
    oa += wj[j] * v[(size_t)(j0 + j) * D_ + h * HD_ + t];
  o[(size_t)r * D_ + h * HD_ + t] = (__bf16)(oa / denom_s);
}

// ------------- relu^2 cross attention, |j - idx[r]| <= 1 mask --------------
__global__ void k_xattn(const float* __restrict__ q, const float* __restrict__ k,
                        const float* __restrict__ v, const int* __restrict__ idx,
                        __bf16* __restrict__ o, int kk) {
  int r = blockIdx.x;
  int h = blockIdx.y;
  int t = threadIdx.x;                       // 0..63
  int slices = (kk >= 64) ? 1 : (HD_ / kk);  // kk 8 -> 8, kk 64 -> 1
  int jj = t % kk;
  int sl = t / kk;
  int w  = HD_ / slices;
  __shared__ float ps[64];
  __shared__ float wj[64];
  __shared__ float denom_s;
  int center = idx[r];
  const float* qrow = q + (size_t)r * D_ + h * HD_;
  const float* krow = k + (size_t)jj * D_ + h * HD_;
  float acc = 0.f;
  for (int i = 0; i < w; ++i) { int d = sl * w + i; acc += qrow[d] * krow[d]; }
  ps[t] = acc;
  __syncthreads();
  if (t < kk) {
    float s = 0.f;
    for (int s2 = 0; s2 < slices; ++s2) s += ps[t + kk * s2];
    float sc = s * 0.125f;
    sc = sc > 0.f ? sc * sc : 0.f;
    int delta = t - center; if (delta < 0) delta = -delta;
    wj[t] = (delta <= 1) ? sc : 0.f;
  }
  __syncthreads();
  if (t == 0) {
    float s = 0.f;
    for (int j = 0; j < kk; ++j) s += wj[j];
    denom_s = s < EPS_ ? EPS_ : s;
  }
  __syncthreads();
  float oa = 0.f;
  for (int j = 0; j < kk; ++j)
    oa += wj[j] * v[(size_t)j * D_ + h * HD_ + t];
  o[(size_t)r * D_ + h * HD_ + t] = (__bf16)(oa / denom_s);
}

// ------------------------------ attention pool -----------------------------
__global__ void k_poolscore(const float* __restrict__ kmat, const float* __restrict__ query,
                            float* __restrict__ s) {
  int r = blockIdx.x;
  __shared__ float red[256];
  float acc = 0.f;
  for (int i = threadIdx.x; i < D_; i += 256) acc += kmat[(size_t)r * D_ + i] * query[i];
  red[threadIdx.x] = acc; __syncthreads();
  for (int st = 128; st > 0; st >>= 1) {
    if ((int)threadIdx.x < st) red[threadIdx.x] += red[threadIdx.x + st];
    __syncthreads();
  }
  if (threadIdx.x == 0) s[r] = red[0] * (1.0f / 32.0f);   // 1/sqrt(1024)
}

__global__ void k_poolout(const float* __restrict__ s, const float* __restrict__ v,
                          float* __restrict__ out, int gsize) {
  int g = blockIdx.x;
  __shared__ float wj[32];
  if (threadIdx.x == 0) {
    float mx = -1e30f;
    for (int j = 0; j < gsize; ++j) mx = fmaxf(mx, s[g * gsize + j]);
    float sum = 0.f;
    for (int j = 0; j < gsize; ++j) { float e = expf(s[g * gsize + j] - mx); wj[j] = e; sum += e; }
    for (int j = 0; j < gsize; ++j) wj[j] /= sum;
  }
  __syncthreads();
  for (int i = threadIdx.x; i < D_; i += blockDim.x) {
    float acc = 0.f;
    for (int j = 0; j < gsize; ++j) acc += wj[j] * v[(size_t)(g * gsize + j) * D_ + i];
    out[(size_t)g * D_ + i] = acc;
  }
}

// ------------------------------ host helpers -------------------------------
static inline void gemm(const __bf16* A, const float* B, const float* Cres, void* Cout,
                        int M, int N, int K, int epi, hipStream_t st) {
  dim3 grid(N / 64, (M + 127) / 128);
  k_gemm<<<grid, 128, 0, st>>>(A, B, Cres, Cout, M, N, K, epi);
}

static void enc_block(void* const* d_in, int pb, float* x, int M,
                      const int* pos, float base, int gsize,
                      __bf16* Hb, float* Qb, float* Kb, float* Vb, __bf16* AOb, __bf16* Fb,
                      hipStream_t st) {
  const float* n1 = (const float*)d_in[pb + 0];
  const float* n2 = (const float*)d_in[pb + 1];
  const float* qn = (const float*)d_in[pb + 2];
  const float* kn = (const float*)d_in[pb + 3];
  const float* wq = (const float*)d_in[pb + 4];
  const float* wk = (const float*)d_in[pb + 5];
  const float* wv = (const float*)d_in[pb + 6];
  const float* wo = (const float*)d_in[pb + 7];
  const float* w1 = (const float*)d_in[pb + 8];
  const float* w2 = (const float*)d_in[pb + 9];
  k_rms_bf16<<<M, 256, 0, st>>>(x, n1, Hb, D_);
  gemm(Hb, wq, nullptr, Qb, M, D_, D_, 0, st);
  gemm(Hb, wk, nullptr, Kb, M, D_, D_, 0, st);
  gemm(Hb, wv, nullptr, Vb, M, D_, D_, 0, st);
  k_qkrope<<<dim3(M, H_), 64, 0, st>>>(Qb, qn, pos, base);
  k_qkrope<<<dim3(M, H_), 64, 0, st>>>(Kb, kn, pos, base);
  k_attn<<<dim3(M, H_), 64, 0, st>>>(Qb, Kb, Vb, AOb, gsize);
  gemm(AOb, wo, x, x, M, D_, D_, 1, st);
  k_rms_bf16<<<M, 256, 0, st>>>(x, n2, Hb, D_);
  gemm(Hb, w1, nullptr, Fb, M, FFN_, D_, 2, st);      // exact GELU, bf16 out
  gemm(Fb, w2, x, x, M, D_, FFN_, 1, st);
}

static void pool_block(void* const* d_in, int pb, const float* x, int M, int gsize, float* out,
                       __bf16* Hb, float* Kmat, float* Vmat, float* Sc, hipStream_t st) {
  const float* query = (const float*)d_in[pb + 0];
  const float* norm  = (const float*)d_in[pb + 1];
  const float* wk    = (const float*)d_in[pb + 2];
  const float* wv    = (const float*)d_in[pb + 3];
  k_rms_bf16<<<M, 256, 0, st>>>(x, norm, Hb, D_);
  gemm(Hb, wk, nullptr, Kmat, M, D_, D_, 0, st);
  gemm(Hb, wv, nullptr, Vmat, M, D_, D_, 0, st);
  k_poolscore<<<M, 256, 0, st>>>(Kmat, query, Sc);
  k_poolout<<<M / gsize, 256, 0, st>>>(Sc, Vmat, out, gsize);
}

static void cross_block(void* const* d_in, int pb, float* x, int M, const float* ctx, int kk,
                        const int* idx, __bf16* Hb, float* Qb, float* Kb, float* Vb,
                        __bf16* AOb, __bf16* Ctxn, hipStream_t st) {
  const float* nq  = (const float*)d_in[pb + 0];
  const float* nkv = (const float*)d_in[pb + 1];
  const float* wq  = (const float*)d_in[pb + 2];
  const float* wk  = (const float*)d_in[pb + 3];
  const float* wv  = (const float*)d_in[pb + 4];
  const float* wo  = (const float*)d_in[pb + 5];
  k_rms_bf16<<<M, 256, 0, st>>>(x, nq, Hb, D_);
  gemm(Hb, wq, nullptr, Qb, M, D_, D_, 0, st);
  k_rms_bf16<<<kk, 256, 0, st>>>(ctx, nkv, Ctxn, D_);
  gemm(Ctxn, wk, nullptr, Kb, kk, D_, D_, 0, st);
  gemm(Ctxn, wv, nullptr, Vb, kk, D_, D_, 0, st);
  k_xattn<<<dim3(M, H_), 64, 0, st>>>(Qb, Kb, Vb, idx, AOb, kk);
  gemm(AOb, wo, x, x, M, D_, D_, 1, st);
}

// ------------------------------ entry point --------------------------------
extern "C" void kernel_launch(void* const* d_in, const int* in_sizes, int n_in,
                              void* d_out, int out_size, void* d_ws, size_t ws_size,
                              hipStream_t stream) {
  (void)in_sizes; (void)n_in; (void)out_size; (void)ws_size;

  // Param index map (setup_inputs() dict insertion order):
  //  0 tok_emb | 1-10 l0[0] | 11-20 l0[1] | 21-30 l1[0] | 31-40 l2[0]
  //  41-44 pool_s | 45-48 pool_p | 49-54 xa_para | 55-60 xa_sent | 61 final_norm
  //  62 token_ids | 63 token_in_sent | 64 sent_id | 65 para_id_per_sent
  //  66 sent_pos_in_para | 67 para_pos
  const float* emb        = (const float*)d_in[0];
  const float* final_norm = (const float*)d_in[61];
  const int* token_ids    = (const int*)d_in[62];
  const int* token_pos    = (const int*)d_in[63];
  const int* sent_id      = (const int*)d_in[64];
  const int* para_id      = (const int*)d_in[65];
  const int* sent_pos     = (const int*)d_in[66];
  const int* para_pos     = (const int*)d_in[67];

  // Workspace layout (floats). bf16 FFN activation overlays Qb+Kb (dead then).
  const size_t BUF = (size_t)T_ * D_;        // 2M floats
  float* ws    = (float*)d_ws;
  float* X     = ws;                         // token activations (persist, f32)
  __bf16* Hb   = (__bf16*)(ws + 1 * BUF);    // rms output, bf16
  float* Qb    = ws + 2 * BUF;
  float* Kb    = ws + 3 * BUF;
  float* Vb    = ws + 4 * BUF;
  __bf16* AOb  = (__bf16*)(ws + 5 * BUF);    // attention output, bf16
  __bf16* Fb   = (__bf16*)(ws + 2 * BUF);    // T_*FFN_ bf16 = spans Qb..Kb
  float* SENT  = ws + 6 * BUF;               // 64*1024
  float* PARA  = SENT + (size_t)NS_ * D_;    // 8*1024
  float* SC    = PARA + (size_t)NP_ * D_;    // up to 2048 scores
  __bf16* CTXN = (__bf16*)(SC + T_);         // 64*1024 ctx rms buffer, bf16

  // 1) embedding gather
  k_embed<<<T_, 256, 0, stream>>>(emb, token_ids, X);

  // 2) token-level encoder blocks (sentence-diagonal mask, gsize=32)
  enc_block(d_in, 1,  X, T_, token_pos, 10000.f, 32, Hb, Qb, Kb, Vb, AOb, Fb, stream);
  enc_block(d_in, 11, X, T_, token_pos, 10000.f, 32, Hb, Qb, Kb, Vb, AOb, Fb, stream);

  // 3) pool tokens -> sentences
  pool_block(d_in, 41, X, T_, 32, SENT, Hb, Qb, Kb, SC, stream);

  // 4) sentence-level encoder (paragraph-diagonal mask, gsize=8)
  enc_block(d_in, 21, SENT, NS_, sent_pos, 1000.f, 8, Hb, Qb, Kb, Vb, AOb, Fb, stream);

  // 5) pool sentences -> paragraphs
  pool_block(d_in, 45, SENT, NS_, 8, PARA, Hb, Qb, Kb, SC, stream);

  // 6) paragraph-level encoder (no mask == one group of 8)
  enc_block(d_in, 31, PARA, NP_, para_pos, 100.f, 8, Hb, Qb, Kb, Vb, AOb, Fb, stream);

  // 7) cross-attention: sentences <- paragraphs, then tokens <- sentences
  cross_block(d_in, 49, SENT, NS_, PARA, NP_, para_id, Hb, Qb, Kb, Vb, AOb, CTXN, stream);
  cross_block(d_in, 55, X,   T_,  SENT, NS_, sent_id, Hb, Qb, Kb, Vb, AOb, CTXN, stream);

  // 8) final RMS norm -> output
  k_rms_f32<<<T_, 256, 0, stream>>>(X, final_norm, (float*)d_out, D_);
}
